// SelfAttnBlock2D_55731495633012
// MI455X (gfx1250) — matrix operand order repaired and checked by reading, not compiled, not run on hardware
//
#include <hip/hip_runtime.h>
#include <hip/hip_bf16.h>

typedef unsigned short u16;
typedef unsigned int   u32;
typedef __attribute__((ext_vector_type(16))) __bf16 v16bf;
typedef __attribute__((ext_vector_type(16))) u16    v16u;
typedef __attribute__((ext_vector_type(8)))  float  v8f;

// ---------------- bf16 helpers (storage = raw u16) ----------------
__device__ __forceinline__ u16 f2bf(float f) {
    u32 u = __builtin_bit_cast(u32, f);
    u32 r = u + 0x7FFFu + ((u >> 16) & 1u);   // round-to-nearest-even
    return (u16)(r >> 16);
}

// D = A(16x32 bf16) * B(32x16 bf16) + C(16x16 f32), wave32 WMMA
__device__ __forceinline__ v8f wmma_bf16(v16u av, v16u bv, v8f c) {
    v16bf a = __builtin_bit_cast(v16bf, av);
    v16bf b = __builtin_bit_cast(v16bf, bv);
    return __builtin_amdgcn_wmma_f32_16x16x32_bf16(
        /*neg_a=*/false, a, /*neg_b=*/false, b,
        /*c_mod=*/(short)0, c, /*reuse_a=*/false, /*reuse_b=*/false);
}

// Async global -> LDS copy of 16 bytes per lane (ASYNCcnt-tracked).
// LDS dest offset = low 32 bits of the generic pointer (LDS aperture maps
// addr[31:0] to the wave's LDS offset). Global address in a 64-bit VGPR pair.
__device__ __forceinline__ void async_b128(void* lds, const void* g) {
    unsigned l = (unsigned)(unsigned long long)(size_t)lds;
    unsigned long long a = (unsigned long long)(size_t)g;
    asm volatile("global_load_async_to_lds_b128 %0, %1, off"
                 :: "v"(l), "v"(a) : "memory");
}
#define WAIT_ASYNCCNT(imm) asm volatile("s_wait_asynccnt " imm ::: "memory")

// A-fragment: src is row-major [M][ld] (K contiguous). 16x32 tile at (row, kbase).
// Lane L: M = row + (L&15); elems 0-7 -> K = kbase + (L>>4)*8 + e
//                           elems 8-15 -> K = kbase + 16 + (L>>4)*8 + (e-8)
__device__ __forceinline__ v16u load_A_frag(const u16* src, int ld, int row, int kbase) {
    int lane = threadIdx.x & 31;
    const u16* p0 = src + (size_t)(row + (lane & 15)) * ld + kbase + ((lane >> 4) << 3);
    const u16* p1 = p0 + 16;
    v16u r;
#pragma unroll
    for (int e = 0; e < 8; ++e) { r[e] = p0[e]; r[e + 8] = p1[e]; }
    return r;
}

// B-fragment: src holds B^T, i.e. row-major [N][ld] with K contiguous.
// Lane L: N = nrow + (L&15); elems 0-7 -> K = kbase + (L>>4)*8 + e, elems 8-15 -> K = kbase + 16 + (L>>4)*8 + (e-8)
__device__ __forceinline__ v16u load_Bt_frag(const u16* src, int ld, int nrow, int kbase) {
    int lane = threadIdx.x & 31;
    const u16* p = src + (size_t)(nrow + (lane & 15)) * ld + kbase + ((lane >> 4) << 3);
    v16u r;
#pragma unroll
    for (int e = 0; e < 8; ++e) { r[e] = p[e]; r[e + 8] = p[e + 16]; }
    return r;
}

// ---------------- geometry ----------------
#define B_    2
#define C_    512
#define SEQ   4096
#define NH    8
#define HD    64
#define MTOT  (B_ * SEQ)      // 8192
#define NQKV  1536

// ---------------- kernel 0: weight/bias prep (fp32 -> bf16) ----------------
__global__ void prep_weights_k(const float* __restrict__ wq, const float* __restrict__ wkv,
                               const float* __restrict__ wproj,
                               const float* __restrict__ bq, const float* __restrict__ bkv,
                               u16* __restrict__ wqkv, u16* __restrict__ wp,
                               float* __restrict__ bias_qkv) {
    int i = blockIdx.x * 256 + threadIdx.x;            // 0 .. 786431
    if (i < NQKV * C_) {
        float v = (i < C_ * C_) ? wq[i] : wkv[i - C_ * C_];
        wqkv[i] = f2bf(v);
    }
    if (i < C_ * C_) wp[i] = f2bf(wproj[i]);
    if (i < NQKV) bias_qkv[i] = (i < C_) ? bq[i] : bkv[i - C_];
}

// ---------------- kernel 1: GroupNorm stats (mean, rstd per (b,group)) ------
__global__ void gn_stats_k(const float* __restrict__ x, float* __restrict__ stats) {
    __shared__ float ssum[256];
    __shared__ float ssq[256];
    int bg = blockIdx.x;                               // b*32 + g, channels contiguous
    const float* base = x + (size_t)bg * 16 * SEQ;     // 16 ch * 4096 pix
    float s = 0.f, q = 0.f;
    for (int i = threadIdx.x; i < 16 * SEQ; i += 256) {
        float v = base[i];
        s += v; q += v * v;
    }
    ssum[threadIdx.x] = s; ssq[threadIdx.x] = q;
    __syncthreads();
    for (int st = 128; st > 0; st >>= 1) {
        if ((int)threadIdx.x < st) {
            ssum[threadIdx.x] += ssum[threadIdx.x + st];
            ssq[threadIdx.x]  += ssq[threadIdx.x + st];
        }
        __syncthreads();
    }
    if (threadIdx.x == 0) {
        float inv = 1.f / (16.f * SEQ);
        float mean = ssum[0] * inv;
        float var  = ssq[0] * inv - mean * mean;
        stats[bg * 2 + 0] = mean;
        stats[bg * 2 + 1] = rsqrtf(var + 1e-5f);
    }
}

// ---------------- kernel 2: GroupNorm apply -> hs bf16 [b*seq, c] -----------
__global__ void gn_apply_k(const float* __restrict__ x, const float* __restrict__ stats,
                           const float* __restrict__ gamma, const float* __restrict__ beta,
                           u16* __restrict__ hs) {
    int bc = blockIdx.x;                               // b*512 + ch
    int b  = bc >> 9, ch = bc & (C_ - 1);
    int g  = ch >> 4;
    float mean = stats[(b * 32 + g) * 2 + 0];
    float rstd = stats[(b * 32 + g) * 2 + 1];
    float w = gamma[ch], bi = beta[ch];
    const float* src = x + (size_t)bc * SEQ;
    u16* dst = hs + (size_t)b * SEQ * C_ + ch;
    for (int p = threadIdx.x; p < SEQ; p += 256) {
        float v = (src[p] - mean) * rstd * w + bi;
        dst[(size_t)p * C_] = f2bf(v);
    }
}

// ---------------- kernel 3: fused QKV GEMM + head reshape -------------------
// C[m,n] = sum_k hs[m,k] * Wqkv[n,k] + bias[n];  M=8192, N=1536, K=512
__global__ __launch_bounds__(128) void qkv_gemm_k(
        const u16* __restrict__ hs, const u16* __restrict__ wqkv,
        const float* __restrict__ bias,
        u16* __restrict__ q, u16* __restrict__ kk, u16* __restrict__ vT) {
    __shared__ __align__(16) u16 As[2][64 * 32];
    __shared__ __align__(16) u16 Bs[2][64 * 32];
    int m0 = blockIdx.x * 64, n0 = blockIdx.y * 64;
    int tid = threadIdx.x, lane = tid & 31, w = tid >> 5;
    int row = tid >> 1, col = (tid & 1) << 4;          // 16 u16 (32B) per thread/array
    const u16* ga = hs   + (size_t)(m0 + row) * C_ + col;
    const u16* gb = wqkv + (size_t)(n0 + row) * C_ + col;

    // prologue: async-stage k0 = 0 into buffer 0
    async_b128(&As[0][row * 32 + col],     ga);
    async_b128(&As[0][row * 32 + col + 8], ga + 8);
    async_b128(&Bs[0][row * 32 + col],     gb);
    async_b128(&Bs[0][row * 32 + col + 8], gb + 8);

    v8f acc[4] = {};
    for (int k0 = 0; k0 < C_; k0 += 32) {
        int buf = (k0 >> 5) & 1;
        if (k0 + 32 < C_) {                            // prefetch next K-step
            const u16* na = ga + k0 + 32;
            const u16* nb = gb + k0 + 32;
            async_b128(&As[buf ^ 1][row * 32 + col],     na);
            async_b128(&As[buf ^ 1][row * 32 + col + 8], na + 8);
            async_b128(&Bs[buf ^ 1][row * 32 + col],     nb);
            async_b128(&Bs[buf ^ 1][row * 32 + col + 8], nb + 8);
            WAIT_ASYNCCNT("0x4");                      // in-order: current tile landed
        } else {
            WAIT_ASYNCCNT("0x0");
        }
        __syncthreads();
        v16u a = load_A_frag(As[buf], 32, w * 16, 0);
#pragma unroll
        for (int t = 0; t < 4; ++t) {
            v16u bf = load_Bt_frag(Bs[buf], 32, t * 16, 0);
            acc[t] = wmma_bf16(a, bf, acc[t]);
        }
        __syncthreads();
    }
#pragma unroll
    for (int t = 0; t < 4; ++t) {
#pragma unroll
        for (int r = 0; r < 8; ++r) {
            int m = m0 + w * 16 + r + ((lane >> 4) << 3);
            int n = n0 + t * 16 + (lane & 15);
            float val = acc[t][r] + bias[n];
            int b = m >> 12, pix = m & (SEQ - 1);
            if (n < C_) {
                int head = n >> 6, d = n & 63;
                q[(((size_t)b * NH + head) * SEQ + pix) * HD + d] = f2bf(val * 0.125f);
            } else if (n < 2 * C_) {
                int nn = n - C_; int head = nn >> 6, d = nn & 63;
                kk[(((size_t)b * NH + head) * SEQ + pix) * HD + d] = f2bf(val);
            } else {
                int nn = n - 2 * C_; int head = nn >> 6, d = nn & 63;
                vT[(((size_t)b * NH + head) * HD + d) * SEQ + pix] = f2bf(val);
            }
        }
    }
}

// ---------------- kernel 4: flash attention per (64 q-rows, batch-head) -----
__global__ __launch_bounds__(128) void attn_k(
        const u16* __restrict__ q, const u16* __restrict__ kmat,
        const u16* __restrict__ vT, u16* __restrict__ attn) {
    __shared__ __align__(16) u16 Ks[2][64 * 64];       // [key][d]
    __shared__ __align__(16) u16 Vs[2][64 * 64];       // [d][key]  (from vT)
    __shared__ __align__(16) u16 Ps[4 * 16 * 64];      // per-wave probs [16][64]
    int qt = blockIdx.x, bh = blockIdx.y;
    int tid = threadIdx.x, lane = tid & 31, w = tid >> 5;
    const size_t bh_off = (size_t)bh * SEQ * HD;

    v16u qf0 = load_A_frag(q + bh_off, HD, qt * 64 + w * 16, 0);
    v16u qf1 = load_A_frag(q + bh_off, HD, qt * 64 + w * 16, 32);

    int row = tid >> 1, col = (tid & 1) << 5;          // 32 u16 (64B) per thread/array
    const u16* gk = kmat + bh_off + (size_t)row * HD + col;   // + kt*64*HD
    const u16* gv = vT   + bh_off + (size_t)row * SEQ + col;  // + kt*64

    auto issue_tile = [&](int kt, int buf) {
        const u16* sk = gk + (size_t)kt * 64 * HD;
        const u16* sv = gv + kt * 64;
#pragma unroll
        for (int j = 0; j < 4; ++j) {
            async_b128(&Ks[buf][row * 64 + col + j * 8], sk + j * 8);
            async_b128(&Vs[buf][row * 64 + col + j * 8], sv + j * 8);
        }
    };
    issue_tile(0, 0);

    float rmax[8], rsum[8];
    v8f oacc[4] = {};
#pragma unroll
    for (int r = 0; r < 8; ++r) { rmax[r] = -1e30f; rsum[r] = 0.f; }

    for (int kt = 0; kt < SEQ / 64; ++kt) {
        int buf = kt & 1;
        if (kt + 1 < SEQ / 64) {                       // prefetch next K/V tile
            issue_tile(kt + 1, buf ^ 1);
            WAIT_ASYNCCNT("0x8");                      // in-order: current tile landed
        } else {
            WAIT_ASYNCCNT("0x0");
        }
        __syncthreads();

        // S = q * K^T  (q pre-scaled by 1/sqrt(hd))
        v8f sacc[4] = {};
#pragma unroll
        for (int ks = 0; ks < 2; ++ks) {
            v16u a = ks ? qf1 : qf0;
#pragma unroll
            for (int t = 0; t < 4; ++t) {
                v16u bf = load_Bt_frag(Ks[buf], 64, t * 16, ks * 32);
                sacc[t] = wmma_bf16(a, bf, sacc[t]);
            }
        }

        // online softmax: row stats per lane (row m = r + 8*(lane>>4)),
        // reduced over the 16-lane column groups with shfl_xor
        float nmax[8], corr[8], tsum[8];
#pragma unroll
        for (int r = 0; r < 8; ++r) {
            float m = fmaxf(fmaxf(sacc[0][r], sacc[1][r]), fmaxf(sacc[2][r], sacc[3][r]));
#pragma unroll
            for (int s = 1; s < 16; s <<= 1) m = fmaxf(m, __shfl_xor(m, s, 32));
            nmax[r] = fmaxf(rmax[r], m);
            corr[r] = __expf(rmax[r] - nmax[r]);
            rmax[r] = nmax[r];
            tsum[r] = 0.f;
        }
#pragma unroll
        for (int t = 0; t < 4; ++t) {
#pragma unroll
            for (int r = 0; r < 8; ++r) {
                float p = __expf(sacc[t][r] - nmax[r]);
                tsum[r] += p;
                // scatter P (C-layout) into row-major [16][64] per-wave staging
                Ps[(w * 16 + r + ((lane >> 4) << 3)) * 64 + t * 16 + (lane & 15)] = f2bf(p);
            }
        }
#pragma unroll
        for (int r = 0; r < 8; ++r) {
            float s = tsum[r];
#pragma unroll
            for (int sm = 1; sm < 16; sm <<= 1) s += __shfl_xor(s, sm, 32);
            rsum[r] = rsum[r] * corr[r] + s;
#pragma unroll
            for (int t = 0; t < 4; ++t) oacc[t][r] *= corr[r];
        }

        // O += P * V   (B operand from Vs[d][k], K contiguous)
#pragma unroll
        for (int ks = 0; ks < 2; ++ks) {
            v16u pa = load_A_frag(Ps, 64, w * 16, ks * 32);
#pragma unroll
            for (int t = 0; t < 4; ++t) {
                v16u bv = load_Bt_frag(Vs[buf], 64, t * 16, ks * 32);
                oacc[t] = wmma_bf16(pa, bv, oacc[t]);
            }
        }
        __syncthreads();
    }

    int b = bh >> 3, head = bh & 7;
#pragma unroll
    for (int t = 0; t < 4; ++t) {
#pragma unroll
        for (int r = 0; r < 8; ++r) {
            int pix = qt * 64 + w * 16 + r + ((lane >> 4) << 3);
            int d = t * 16 + (lane & 15);
            float val = oacc[t][r] / rsum[r];
            attn[((size_t)b * SEQ + pix) * C_ + head * HD + d] = f2bf(val);
        }
    }
}

// ---------------- kernel 5: out projection + bias + residual ----------------
__global__ __launch_bounds__(128) void proj_gemm_k(
        const u16* __restrict__ attn, const u16* __restrict__ wp,
        const float* __restrict__ bproj, const float* __restrict__ x,
        float* __restrict__ out) {
    __shared__ __align__(16) u16 As[2][64 * 32];
    __shared__ __align__(16) u16 Bs[2][64 * 32];
    int m0 = blockIdx.x * 64, n0 = blockIdx.y * 64;
    int tid = threadIdx.x, lane = tid & 31, w = tid >> 5;
    int row = tid >> 1, col = (tid & 1) << 4;
    const u16* ga = attn + (size_t)(m0 + row) * C_ + col;
    const u16* gb = wp   + (size_t)(n0 + row) * C_ + col;

    async_b128(&As[0][row * 32 + col],     ga);
    async_b128(&As[0][row * 32 + col + 8], ga + 8);
    async_b128(&Bs[0][row * 32 + col],     gb);
    async_b128(&Bs[0][row * 32 + col + 8], gb + 8);

    v8f acc[4] = {};
    for (int k0 = 0; k0 < C_; k0 += 32) {
        int buf = (k0 >> 5) & 1;
        if (k0 + 32 < C_) {
            const u16* na = ga + k0 + 32;
            const u16* nb = gb + k0 + 32;
            async_b128(&As[buf ^ 1][row * 32 + col],     na);
            async_b128(&As[buf ^ 1][row * 32 + col + 8], na + 8);
            async_b128(&Bs[buf ^ 1][row * 32 + col],     nb);
            async_b128(&Bs[buf ^ 1][row * 32 + col + 8], nb + 8);
            WAIT_ASYNCCNT("0x4");
        } else {
            WAIT_ASYNCCNT("0x0");
        }
        __syncthreads();
        v16u a = load_A_frag(As[buf], 32, w * 16, 0);
#pragma unroll
        for (int t = 0; t < 4; ++t) {
            v16u bf = load_Bt_frag(Bs[buf], 32, t * 16, 0);
            acc[t] = wmma_bf16(a, bf, acc[t]);
        }
        __syncthreads();
    }
#pragma unroll
    for (int t = 0; t < 4; ++t) {
#pragma unroll
        for (int r = 0; r < 8; ++r) {
            int m = m0 + w * 16 + r + ((lane >> 4) << 3);
            int n = n0 + t * 16 + (lane & 15);
            int b = m >> 12, pix = m & (SEQ - 1);
            size_t xi = ((size_t)b * C_ + n) * SEQ + pix;  // [b,c,h,w] layout
            out[xi] = acc[t][r] + bproj[n] + x[xi];
        }
    }
}

// ---------------- host side -------------------------------------------------
extern "C" void kernel_launch(void* const* d_in, const int* in_sizes, int n_in,
                              void* d_out, int out_size, void* d_ws, size_t ws_size,
                              hipStream_t stream) {
    (void)in_sizes; (void)n_in; (void)out_size; (void)ws_size;
    const float* x     = (const float*)d_in[0];
    const float* gw    = (const float*)d_in[1];
    const float* gb    = (const float*)d_in[2];
    const float* wq    = (const float*)d_in[3];
    const float* bq    = (const float*)d_in[4];
    const float* wkv   = (const float*)d_in[5];
    const float* bkv   = (const float*)d_in[6];
    const float* wproj = (const float*)d_in[7];
    const float* bproj = (const float*)d_in[8];
    float* out = (float*)d_out;

    char* ws = (char*)d_ws;
    u16*   wqkv_bf  = (u16*)(ws + 0x0000000);   // 1536*512*2 = 0x180000
    u16*   wp_bf    = (u16*)(ws + 0x0180000);   // 512*512*2  = 0x080000
    float* bias_qkv = (float*)(ws + 0x0200000); // 1536*4
    float* stats    = (float*)(ws + 0x0202000); // 64*2*4
    u16*   hs       = (u16*)(ws + 0x0210000);   // 8192*512*2 = 0x800000
    u16*   qbuf     = (u16*)(ws + 0x0A10000);   // 0x800000
    u16*   kbuf     = (u16*)(ws + 0x1210000);   // 0x800000
    u16*   vbuf     = (u16*)(ws + 0x1A10000);   // 0x800000
    u16*   attnbuf  = (u16*)(ws + 0x2210000);   // 0x800000  (total ~44.1 MB)

    prep_weights_k<<<(NQKV * C_) / 256, 256, 0, stream>>>(wq, wkv, wproj, bq, bkv,
                                                          wqkv_bf, wp_bf, bias_qkv);
    gn_stats_k<<<B_ * 32, 256, 0, stream>>>(x, stats);
    gn_apply_k<<<B_ * C_, 256, 0, stream>>>(x, stats, gw, gb, hs);
    qkv_gemm_k<<<dim3(MTOT / 64, NQKV / 64), 128, 0, stream>>>(hs, wqkv_bf, bias_qkv,
                                                               qbuf, kbuf, vbuf);
    attn_k<<<dim3(SEQ / 64, B_ * NH), 128, 0, stream>>>(qbuf, kbuf, vbuf, attnbuf);
    proj_gemm_k<<<dim3(MTOT / 64, C_ / 64), 128, 0, stream>>>(attnbuf, wp_bf, bproj, x, out);
}
